// MyGCNNet_18459769438305
// MI455X (gfx1250) — matrix operand order repaired and checked by reading, not compiled, not run on hardware
//
#include <hip/hip_runtime.h>
#include <math.h>

// ---------------- model dims ----------------
#define BB   16
#define HWI  224
#define HC   56
#define SPD  14
#define N1C  (BB*HC*HC)        // 50176 nodes level 1
#define E1C  (9*N1C)           // 451584 edges level 1
#define N2C  (BB*SPD*SPD)      // 3136 nodes level 2
#define E2C  (17*N2C)          // 53312 edges level 2
#define DD   128

typedef __attribute__((ext_vector_type(16))) _Float16 v16h;
typedef __attribute__((ext_vector_type(8)))  _Float16 h8;
typedef __attribute__((ext_vector_type(8)))  float    v8f;

__device__ __forceinline__ int iclamp(int v, int lo, int hi) {
  return v < lo ? lo : (v > hi ? hi : v);
}

// =====================================================================
// weight packers (run once per launch; tiny)
// =====================================================================
__global__ void pack_conv_w(const float* __restrict__ w, _Float16* __restrict__ o,
                            int Cout, int K, int Kp) {
  int idx = blockIdx.x * blockDim.x + threadIdx.x;
  if (idx >= Cout * Kp) return;
  int n = idx / Kp, k = idx % Kp;
  o[idx] = (k < K) ? (_Float16)w[(size_t)n * K + k] : (_Float16)0.f;
}
// conv OIHW f32 -> f16 [Cout x Cin*9], K order = r*Cin+c (tap-major)
__global__ void pack_conv_w_rmaj(const float* __restrict__ w,
                                 _Float16* __restrict__ o, int Cout, int Cin) {
  int K = Cin * 9;
  int idx = blockIdx.x * blockDim.x + threadIdx.x;
  if (idx >= Cout * K) return;
  int n = idx / K, k = idx % K;
  int r = k / Cin, c = k % Cin;
  o[idx] = (_Float16)w[((size_t)n * Cin + c) * 9 + r];
}
// GEMM W (K x N) f32 -> f16 transposed (N x K)
__global__ void pack_gemm_w(const float* __restrict__ w, _Float16* __restrict__ o,
                            int K, int N) {
  int idx = blockIdx.x * blockDim.x + threadIdx.x;
  if (idx >= N * K) return;
  int n = idx / K, k = idx % K;
  o[idx] = (_Float16)w[(size_t)k * N + n];
}

// ---- shared fragment helpers ----
#define LOADB(bv, wr, ko)                                         \
    { h8 lo = *reinterpret_cast<const h8*>((wr) + (ko));          \
      h8 hi = *reinterpret_cast<const h8*>((wr) + (ko) + 16);     \
      _Pragma("unroll")                                           \
      for (int e = 0; e < 8; ++e) { bv[e] = lo[e]; bv[e+8] = hi[e]; } }

__device__ __forceinline__ v16h load_a_f32(const float* Ap, int half) {
  float4 f0 = *reinterpret_cast<const float4*>(Ap + half * 8);
  float4 f1 = *reinterpret_cast<const float4*>(Ap + half * 8 + 4);
  float4 f2 = *reinterpret_cast<const float4*>(Ap + 16 + half * 8);
  float4 f3 = *reinterpret_cast<const float4*>(Ap + 16 + half * 8 + 4);
  v16h a;
  a[0]=(_Float16)f0.x;  a[1]=(_Float16)f0.y;  a[2]=(_Float16)f0.z;  a[3]=(_Float16)f0.w;
  a[4]=(_Float16)f1.x;  a[5]=(_Float16)f1.y;  a[6]=(_Float16)f1.z;  a[7]=(_Float16)f1.w;
  a[8]=(_Float16)f2.x;  a[9]=(_Float16)f2.y;  a[10]=(_Float16)f2.z; a[11]=(_Float16)f2.w;
  a[12]=(_Float16)f3.x; a[13]=(_Float16)f3.y; a[14]=(_Float16)f3.z; a[15]=(_Float16)f3.w;
  return a;
}

// =====================================================================
// WMMA GEMM: out[M,NN] = concat(A1[M,K1],A2[M,K2]) @ W + bias
// W pre-packed f16 as (NN x K) row-major. K1/K2/NN compile-time.
// 16x64 tile/wave; grid = (M/16, ceil(NN/64)); block 32.
// =====================================================================
template <int K1, int K2, int NN>
__global__ void gemm_f16w(const float* __restrict__ A1,
                          const float* __restrict__ A2,
                          const _Float16* __restrict__ Wt,
                          const float* __restrict__ bias,
                          float* __restrict__ out, int M) {
  const int lane = threadIdx.x & 31;
  const int col  = lane & 15;
  const int half = lane >> 4;
  const int m0   = blockIdx.x * 16;
  const int nb   = blockIdx.y * 64;
  constexpr int K = K1 + K2;
  const int m    = m0 + col;

  const _Float16* wr0 = Wt + (size_t)iclamp(nb +      col, 0, NN - 1) * K;
  const _Float16* wr1 = Wt + (size_t)iclamp(nb + 16 + col, 0, NN - 1) * K;
  const _Float16* wr2 = Wt + (size_t)iclamp(nb + 32 + col, 0, NN - 1) * K;
  const _Float16* wr3 = Wt + (size_t)iclamp(nb + 48 + col, 0, NN - 1) * K;
  const float* a1row = A1 + (size_t)m * K1;
  const float* a2row = A2 + (size_t)m * K2;

  v8f acc0 = {}, acc1 = {}, acc2 = {}, acc3 = {};
#pragma unroll
  for (int k0 = 0; k0 < K; k0 += 32) {
    const float* Ap = (k0 < K1) ? (a1row + k0) : (a2row + (k0 - K1));
    v16h a = load_a_f32(Ap, half);
    const int ko = k0 + half * 8;
    v16h b0, b1, b2, b3;
    LOADB(b0, wr0, ko) LOADB(b1, wr1, ko) LOADB(b2, wr2, ko) LOADB(b3, wr3, ko)
    acc0 = __builtin_amdgcn_wmma_f32_16x16x32_f16(false, a, false, b0, (short)0, acc0, false, false);
    acc1 = __builtin_amdgcn_wmma_f32_16x16x32_f16(false, a, false, b1, (short)0, acc1, false, false);
    acc2 = __builtin_amdgcn_wmma_f32_16x16x32_f16(false, a, false, b2, (short)0, acc2, false, false);
    acc3 = __builtin_amdgcn_wmma_f32_16x16x32_f16(false, a, false, b3, (short)0, acc3, false, false);
  }

  const int mrow = m0 + half * 8;
#define STORE(accj, j)                                                      \
  { int n = nb + (j)*16 + col;                                              \
    if (n < NN) { float bv = bias ? bias[n] : 0.f;                          \
      _Pragma("unroll")                                                     \
      for (int r = 0; r < 8; ++r)                                           \
        out[(size_t)(mrow + r) * NN + n] = accj[r] + bv; } }
  STORE(acc0, 0) STORE(acc1, 1) STORE(acc2, 2) STORE(acc3, 3)
#undef STORE
}

// ---- vectorized conv store epilogue: rows of one lane are 8 consecutive
// pixels of a channel plane -> two b128 stores per n-tile (HW % 16 == 0) ----
#define CONV_STORE(accj, j)                                                   \
  { int n = nb + (j)*16 + col;                                                \
    float bv = bias[n];                                                       \
    float* yp = y + ((size_t)bimgs * Cout + n) * HW + re0 + half * 8;         \
    float4 s0, s1;                                                            \
    s0.x = accj[0] + bv; s0.y = accj[1] + bv;                                 \
    s0.z = accj[2] + bv; s0.w = accj[3] + bv;                                 \
    s1.x = accj[4] + bv; s1.y = accj[5] + bv;                                 \
    s1.z = accj[6] + bv; s1.w = accj[7] + bv;                                 \
    *reinterpret_cast<float4*>(yp)     = s0;                                  \
    *reinterpret_cast<float4*>(yp + 4) = s1; }

// =====================================================================
// FAST implicit-GEMM 3x3 conv (pad 1), NCHW. CIN % 32 == 0.
// Weights pre-packed f16 [Cout x CIN*9] in tap-major order (kk = r*CIN + c):
// each 32-wide k-step lies inside one tap -> scalar (r, cbase), uniform
// boundary predicate, A loads = one base pointer + immediate offsets.
// grid = (M/16, Cout/64), block 32.
// =====================================================================
template <int CIN, int H, int Wd>
__global__ void conv3x3_fast(const float* __restrict__ x,
                             const _Float16* __restrict__ wp,
                             const float* __restrict__ bias,
                             float* __restrict__ y, int Cout) {
  constexpr int HW = H * Wd;
  constexpr int K  = CIN * 9;
  const int lane = threadIdx.x & 31;
  const int col  = lane & 15;
  const int half = lane >> 4;
  const int m0   = blockIdx.x * 16;
  const int nb   = blockIdx.y * 64;
  const int bimgs = m0 / HW;         // tile-uniform (HW % 16 == 0)
  const int re0   = m0 % HW;
  const int rem   = re0 + col;
  const int oy    = rem / Wd;
  const int ox    = rem % Wd;
  const float* xb = x + (size_t)bimgs * CIN * HW + (size_t)half * 8 * HW;

  const _Float16* wr0 = wp + (size_t)(nb +      col) * K;
  const _Float16* wr1 = wp + (size_t)(nb + 16 + col) * K;
  const _Float16* wr2 = wp + (size_t)(nb + 32 + col) * K;
  const _Float16* wr3 = wp + (size_t)(nb + 48 + col) * K;

  v8f acc0 = {}, acc1 = {}, acc2 = {}, acc3 = {};
#pragma unroll 2
  for (int k0 = 0; k0 < K; k0 += 32) {
    const int r     = k0 / CIN;          // scalar: tap id
    const int cbase = k0 % CIN;          // scalar: channel base
    const int dy = r / 3 - 1, dx = r % 3 - 1;
    const int iy = oy + dy, ix = ox + dx;
    const bool ok = (iy >= 0) & (iy < H) & (ix >= 0) & (ix < Wd);
    const int iyc = iclamp(iy, 0, H - 1), ixc = iclamp(ix, 0, Wd - 1);
    const float* p = xb + (size_t)cbase * HW + iyc * Wd + ixc;

    v16h a;
#pragma unroll
    for (int e = 0; e < 16; ++e) {
      const int coff = (e < 8) ? e : (e + 8);       // channel offset 0..7,16..23
      float ld = p[(size_t)coff * HW];              // immediate-offset load
      a[e] = (_Float16)(ok ? ld : 0.f);
    }

    const int ko = k0 + half * 8;
    v16h b0, b1, b2, b3;
    LOADB(b0, wr0, ko) LOADB(b1, wr1, ko) LOADB(b2, wr2, ko) LOADB(b3, wr3, ko)
    acc0 = __builtin_amdgcn_wmma_f32_16x16x32_f16(false, a, false, b0, (short)0, acc0, false, false);
    acc1 = __builtin_amdgcn_wmma_f32_16x16x32_f16(false, a, false, b1, (short)0, acc1, false, false);
    acc2 = __builtin_amdgcn_wmma_f32_16x16x32_f16(false, a, false, b2, (short)0, acc2, false, false);
    acc3 = __builtin_amdgcn_wmma_f32_16x16x32_f16(false, a, false, b3, (short)0, acc3, false, false);
  }

  CONV_STORE(acc0, 0) CONV_STORE(acc1, 1) CONV_STORE(acc2, 2) CONV_STORE(acc3, 3)
}

// =====================================================================
// generic conv (used only for conv1: CIN=3, K=27 padded to 32)
// =====================================================================
template <int CIN, int KK, int KP, int H, int Wd>
__global__ void conv3x3_gen(const float* __restrict__ x,
                            const _Float16* __restrict__ wp,
                            const float* __restrict__ bias,
                            float* __restrict__ y, int Cout) {
  constexpr int HW = H * Wd;
  const int lane = threadIdx.x & 31;
  const int col  = lane & 15;
  const int half = lane >> 4;
  const int m0   = blockIdx.x * 16;
  const int nb   = blockIdx.y * 64;
  const int bimgs = m0 / HW;
  const int re0   = m0 % HW;
  const int rem   = re0 + col;
  const int oy    = rem / Wd;
  const int ox    = rem % Wd;
  const float* xb = x + (size_t)bimgs * CIN * HW;

  const _Float16* wr0 = wp + (size_t)(nb +      col) * KP;
  const _Float16* wr1 = wp + (size_t)(nb + 16 + col) * KP;
  const _Float16* wr2 = wp + (size_t)(nb + 32 + col) * KP;
  const _Float16* wr3 = wp + (size_t)(nb + 48 + col) * KP;

  v8f acc0 = {}, acc1 = {}, acc2 = {}, acc3 = {};
#pragma unroll 2
  for (int k0 = 0; k0 < KP; k0 += 32) {
    v16h a;
#pragma unroll
    for (int e = 0; e < 16; ++e) {
      int ka  = k0 + e + (e < 8 ? 0 : 8) + (half ? 8 : 0);
      int kac = ka < KK ? ka : KK - 1;
      int c = kac / 9, r = kac % 9;
      int iy = oy + r / 3 - 1, ix = ox + r % 3 - 1;
      bool ok = (ka < KK) & (iy >= 0) & (iy < H) & (ix >= 0) & (ix < Wd);
      int iyc = iclamp(iy, 0, H - 1), ixc = iclamp(ix, 0, Wd - 1);
      float ld = xb[((size_t)c * H + iyc) * Wd + ixc];
      a[e] = (_Float16)(ok ? ld : 0.f);
    }
    const int ko = k0 + half * 8;
    v16h b0, b1, b2, b3;
    LOADB(b0, wr0, ko) LOADB(b1, wr1, ko) LOADB(b2, wr2, ko) LOADB(b3, wr3, ko)
    acc0 = __builtin_amdgcn_wmma_f32_16x16x32_f16(false, a, false, b0, (short)0, acc0, false, false);
    acc1 = __builtin_amdgcn_wmma_f32_16x16x32_f16(false, a, false, b1, (short)0, acc1, false, false);
    acc2 = __builtin_amdgcn_wmma_f32_16x16x32_f16(false, a, false, b2, (short)0, acc2, false, false);
    acc3 = __builtin_amdgcn_wmma_f32_16x16x32_f16(false, a, false, b3, (short)0, acc3, false, false);
  }

  CONV_STORE(acc0, 0) CONV_STORE(acc1, 1) CONV_STORE(acc2, 2) CONV_STORE(acc3, 3)
}

// ---------------- BatchNorm2d stats: one block per channel ----------------
__global__ void bn2d_stats(const float* __restrict__ x, float* mean, float* istd,
                           int C, int Bn, int HW2) {
  const int c = blockIdx.x;
  __shared__ float s1[256], s2[256];
  float a = 0.f, b = 0.f;
  for (int bi = 0; bi < Bn; ++bi) {
    const float* plane = x + ((size_t)bi * C + c) * HW2;
    for (int j = threadIdx.x; j < HW2; j += blockDim.x) {
      float v = plane[j];
      a += v; b += v * v;
    }
  }
  s1[threadIdx.x] = a; s2[threadIdx.x] = b; __syncthreads();
  for (int s = 128; s > 0; s >>= 1) {
    if (threadIdx.x < s) { s1[threadIdx.x] += s1[threadIdx.x + s];
                           s2[threadIdx.x] += s2[threadIdx.x + s]; }
    __syncthreads();
  }
  if (threadIdx.x == 0) {
    float tot = (float)Bn * (float)HW2;
    float mu = s1[0] / tot;
    float va = s2[0] / tot - mu * mu;
    mean[c] = mu;
    istd[c] = rsqrtf(va + 1e-5f);
  }
}

template <int C, int HW2>
__global__ void bn2d_apply_relu(float* __restrict__ x, const float* mean,
                                const float* istd, const float* g, const float* b,
                                size_t total) {
  size_t idx = (size_t)blockIdx.x * blockDim.x + threadIdx.x;
  if (idx >= total) return;
  int c = (int)((idx / HW2) % C);
  float v = (x[idx] - mean[c]) * istd[c] * g[c] + b[c];
  x[idx] = fmaxf(v, 0.f);
}

// ---------------- 2x2 max pool (H/W compile-time) ----------------
template <int H, int W>
__global__ void maxpool2(const float* __restrict__ x, float* __restrict__ y,
                         int BC) {
  constexpr int OH = H >> 1, OW = W >> 1;
  size_t total = (size_t)BC * OH * OW;
  size_t idx = (size_t)blockIdx.x * blockDim.x + threadIdx.x;
  if (idx >= total) return;
  int ow = (int)(idx % OW);
  int oh = (int)((idx / OW) % OH);
  int p  = (int)(idx / ((size_t)OW * OH));
  const float* base = x + ((size_t)p * H + 2 * oh) * W + 2 * ow;
  y[idx] = fmaxf(fmaxf(base[0], base[1]), fmaxf(base[W], base[W + 1]));
}

// ---------------- gather pixel features (NCHW -> node major) --------------
__global__ void gather_pix(const float* __restrict__ f, const int* __restrict__ wh,
                           float* __restrict__ pix) {
  size_t idx = (size_t)blockIdx.x * blockDim.x + threadIdx.x;
  if (idx >= (size_t)N1C * DD) return;
  int c = (int)(idx % DD);
  int i = (int)(idx / DD);
  int b = wh[i * 3 + 0], r = wh[i * 3 + 1], cc = wh[i * 3 + 2];
  pix[idx] = f[(((size_t)b * DD + c) * HC + r) * HC + cc];
}

// ---------------- edge MLP + message scatter (one block per edge) ---------
__global__ void edge_msg_aggr(const float* __restrict__ x,
                              const int* __restrict__ ei,
                              const float* __restrict__ ew,
                              const float* __restrict__ w1,   // (2,64)
                              const float* __restrict__ w2,   // (64,128)
                              float* __restrict__ aggr, float* __restrict__ cnt,
                              int E) {
  const int e = blockIdx.x;
  const int t = threadIdx.x;
  __shared__ float h1[64];
  float e0 = ew[(size_t)e * 2 + 0];
  float e1 = ew[(size_t)e * 2 + 1];
  if (t < 64) h1[t] = fmaxf(e0 * w1[t] + e1 * w1[64 + t], 0.f);
  __syncthreads();
  float pe = 0.f;
#pragma unroll 8
  for (int k = 0; k < 64; ++k) pe += h1[k] * w2[(size_t)k * DD + t];
  int src = ei[e];
  int dst = ei[(size_t)E + e];
  float xj = x[(size_t)src * DD + t];
  atomicAdd(&aggr[(size_t)dst * DD + t], pe * xj + xj);
  if (t == 0) atomicAdd(&cnt[dst], 1.f);
}

__global__ void seg_div(float* __restrict__ aggr, const float* __restrict__ cnt,
                        int Nn) {
  size_t idx = (size_t)blockIdx.x * blockDim.x + threadIdx.x;
  if (idx >= (size_t)Nn * DD) return;
  float d = cnt[idx / DD];
  aggr[idx] /= fmaxf(d, 1.f);
}

// ---------------- per-row L2 normalize (block of 128 per row) -------------
__global__ void rownorm(const float* __restrict__ h, float* __restrict__ hn) {
  const int i = blockIdx.x, t = threadIdx.x;
  __shared__ float s[128];
  float v = h[(size_t)i * DD + t];
  s[t] = v * v; __syncthreads();
  for (int k = 64; k > 0; k >>= 1) { if (t < k) s[t] += s[t + k]; __syncthreads(); }
  float nrm = fmaxf(sqrtf(s[0]), 1e-12f);
  hn[(size_t)i * DD + t] = v / nrm;
}

// ---------------- BatchNorm1d stats: one block per channel ----------------
__global__ void colstats(const float* __restrict__ h, float* mean, float* istd,
                         int M) {
  const int c = blockIdx.x;
  __shared__ float s1[256], s2[256];
  float a = 0.f, b = 0.f;
  for (int i = threadIdx.x; i < M; i += blockDim.x) {
    float v = h[(size_t)i * DD + c]; a += v; b += v * v;
  }
  s1[threadIdx.x] = a; s2[threadIdx.x] = b; __syncthreads();
  for (int s = 128; s > 0; s >>= 1) {
    if (threadIdx.x < s) { s1[threadIdx.x] += s1[threadIdx.x + s];
                           s2[threadIdx.x] += s2[threadIdx.x + s]; }
    __syncthreads();
  }
  if (threadIdx.x == 0) {
    float mu = s1[0] / M;
    float va = s2[0] / M - mu * mu;
    mean[c] = mu;
    istd[c] = rsqrtf(va + 1e-5f);
  }
}

__global__ void bn1d_apply_relu_res(const float* __restrict__ hn,
                                    const float* mean, const float* istd,
                                    const float* g, const float* b,
                                    const float* __restrict__ xres,
                                    float* __restrict__ out, size_t total) {
  size_t idx = (size_t)blockIdx.x * blockDim.x + threadIdx.x;
  if (idx >= total) return;
  int c = (int)(idx % DD);
  float v = (hn[idx] - mean[c]) * istd[c] * g[c] + b[c];
  out[idx] = fmaxf(v + xres[idx], 0.f);
}

// ---------------- segment mean accumulate (block of 128 per row) ----------
__global__ void seg_accum(const float* __restrict__ x, const int* __restrict__ seg,
                          float* __restrict__ aggr, float* __restrict__ cnt) {
  const int i = blockIdx.x, t = threadIdx.x;
  const int s = seg[i];
  atomicAdd(&aggr[(size_t)s * DD + t], x[(size_t)i * DD + t]);
  if (t == 0) atomicAdd(&cnt[s], 1.f);
}

// ---------------- attention gate: g = g*(1+sigmoid(g@aw))/2 ---------------
__global__ void attn_scale(float* __restrict__ g, const float* __restrict__ aw) {
  const int i = blockIdx.x, t = threadIdx.x;
  __shared__ float s[128];
  float v = g[(size_t)i * DD + t];
  s[t] = v * aw[t]; __syncthreads();
  for (int k = 64; k > 0; k >>= 1) { if (t < k) s[t] += s[t + k]; __syncthreads(); }
  float att = 1.f / (1.f + expf(-s[0]));
  g[(size_t)i * DD + t] = v * (1.f + att) * 0.5f;
}

// ---------------- segment max ----------------
__global__ void init_val(float* __restrict__ p, int n, float v) {
  int i = blockIdx.x * blockDim.x + threadIdx.x;
  if (i < n) p[i] = v;
}

__device__ __forceinline__ void atomicMaxF(float* addr, float v) {
  unsigned int* ai = (unsigned int*)addr;
  unsigned int old = *ai;
  while (true) {
    float f = __uint_as_float(old);
    if (f >= v) break;
    unsigned int prev = atomicCAS(ai, old, __float_as_uint(v));
    if (prev == old) break;
    old = prev;
  }
}

__global__ void seg_max(const float* __restrict__ x, const int* __restrict__ seg,
                        float* __restrict__ out) {
  const int i = blockIdx.x, t = threadIdx.x;
  atomicMaxF(&out[(size_t)seg[i] * DD + t], x[(size_t)i * DD + t]);
}

// =====================================================================
//                            orchestration
// =====================================================================
static inline unsigned cdiv(size_t a, unsigned b) { return (unsigned)((a + b - 1) / b); }

extern "C" void kernel_launch(void* const* d_in, const int* in_sizes, int n_in,
                              void* d_out, int out_size, void* d_ws, size_t ws_size,
                              hipStream_t stream) {
  auto F = [&](int i) { return (const float*)d_in[i]; };
  auto I = [&](int i) { return (const int*)d_in[i]; };

  const float* images = F(0);
  const int*   ei1    = I(2);
  const float* ew1    = F(3);
  const int*   batch1 = I(4);
  const int*   ei2    = I(5);
  const float* ew2    = F(6);
  const int*   batch2 = I(7);
  // params (dict-order leaves), base 8:
  // 8..23 conv/bn, 24 emb1_w, 25..36 blocks1 (2x6), 37 emb2_w,
  // 38..61 blocks2 (4x6), 62 attn_w, 63 readout_w

  float* ws = (float*)d_ws;
  const size_t BIG = (size_t)BB * 64 * HWI * HWI;  // 51,380,224 floats
  float* P0 = ws;
  float* P1 = ws + BIG;
  float* SC = ws + 2 * BIG;

  float* mean = SC;
  float* istd = SC + 128;
  float* cnt1 = SC + 256;                      // N1C floats
  _Float16* PW = (_Float16*)(SC + 256 + N1C); // packed-f16 weight arena

  _Float16* pw_c1   = PW;                    // 64  x 32
  _Float16* pw_c2   = pw_c1 + 64 * 32;       // 64  x 576
  _Float16* pw_c3   = pw_c2 + 64 * 576;      // 128 x 576
  _Float16* pw_c4   = pw_c3 + 128 * 576;     // 128 x 1152
  _Float16* pw_emb1 = pw_c4 + 128 * 1152;    // 128 x 128
  _Float16* pw_l1a  = pw_emb1 + 128 * 128;   // 128 x 256
  _Float16* pw_l1b  = pw_l1a + 128 * 256;
  _Float16* pw_emb2 = pw_l1b + 128 * 256;    // 128 x 128
  _Float16* pw_l2[4];
  pw_l2[0] = pw_emb2 + 128 * 128;
  pw_l2[1] = pw_l2[0] + 128 * 256;
  pw_l2[2] = pw_l2[1] + 128 * 256;
  pw_l2[3] = pw_l2[2] + 128 * 256;
  _Float16* pw_ro = pw_l2[3] + 128 * 256;    // 1000 x 128

  // ---- pack all weights to f16 ----
  pack_conv_w<<<cdiv(64 * 32, 256), 256, 0, stream>>>(F(8), pw_c1, 64, 27, 32);
  pack_conv_w_rmaj<<<cdiv(64 * 576, 256),   256, 0, stream>>>(F(12), pw_c2, 64, 64);
  pack_conv_w_rmaj<<<cdiv(128 * 576, 256),  256, 0, stream>>>(F(16), pw_c3, 128, 64);
  pack_conv_w_rmaj<<<cdiv(128 * 1152, 256), 256, 0, stream>>>(F(20), pw_c4, 128, 128);
  pack_gemm_w<<<cdiv(128 * 128, 256),  256, 0, stream>>>(F(24), pw_emb1, 128, 128);
  pack_gemm_w<<<cdiv(128 * 256, 256),  256, 0, stream>>>(F(27), pw_l1a, 256, 128);
  pack_gemm_w<<<cdiv(128 * 256, 256),  256, 0, stream>>>(F(33), pw_l1b, 256, 128);
  pack_gemm_w<<<cdiv(128 * 128, 256),  256, 0, stream>>>(F(37), pw_emb2, 128, 128);
  for (int i = 0; i < 4; ++i)
    pack_gemm_w<<<cdiv(128 * 256, 256), 256, 0, stream>>>(F(38 + i * 6 + 2), pw_l2[i], 256, 128);
  pack_gemm_w<<<cdiv(1000 * 128, 256), 256, 0, stream>>>(F(63), pw_ro, 128, 1000);

  const size_t NB  = (size_t)N1C * DD;
  const size_t NB2 = (size_t)N2C * DD;

  // ---------------- conv stack ----------------
  {
    size_t M = (size_t)BB * HWI * HWI;                       // 802816
    conv3x3_gen<3, 27, 32, HWI, HWI><<<dim3((unsigned)(M / 16), 1), 32, 0, stream>>>(
        images, pw_c1, F(9), P0, 64);
    bn2d_stats<<<64, 256, 0, stream>>>(P0, mean, istd, 64, BB, HWI * HWI);
    bn2d_apply_relu<64, HWI * HWI><<<cdiv(M * 64, 256), 256, 0, stream>>>(
        P0, mean, istd, F(10), F(11), M * 64);

    conv3x3_fast<64, HWI, HWI><<<dim3((unsigned)(M / 16), 1), 32, 0, stream>>>(
        P0, pw_c2, F(13), P1, 64);
    bn2d_stats<<<64, 256, 0, stream>>>(P1, mean, istd, 64, BB, HWI * HWI);
    bn2d_apply_relu<64, HWI * HWI><<<cdiv(M * 64, 256), 256, 0, stream>>>(
        P1, mean, istd, F(14), F(15), M * 64);

    maxpool2<HWI, HWI><<<cdiv((size_t)BB * 64 * 112 * 112, 256), 256, 0, stream>>>(
        P1, P0, BB * 64);

    size_t M2 = (size_t)BB * 112 * 112;                      // 200704
    conv3x3_fast<64, 112, 112><<<dim3((unsigned)(M2 / 16), 2), 32, 0, stream>>>(
        P0, pw_c3, F(17), P1, 128);
    bn2d_stats<<<128, 256, 0, stream>>>(P1, mean, istd, 128, BB, 112 * 112);
    bn2d_apply_relu<128, 112 * 112><<<cdiv(M2 * 128, 256), 256, 0, stream>>>(
        P1, mean, istd, F(18), F(19), M2 * 128);

    conv3x3_fast<128, 112, 112><<<dim3((unsigned)(M2 / 16), 2), 32, 0, stream>>>(
        P1, pw_c4, F(21), P0, 128);
    bn2d_stats<<<128, 256, 0, stream>>>(P0, mean, istd, 128, BB, 112 * 112);
    bn2d_apply_relu<128, 112 * 112><<<cdiv(M2 * 128, 256), 256, 0, stream>>>(
        P0, mean, istd, F(22), F(23), M2 * 128);

    maxpool2<112, 112><<<cdiv((size_t)BB * 128 * HC * HC, 256), 256, 0, stream>>>(
        P0, P1, BB * 128);
  }

  // ---------------- gather node features ----------------
  float* x0   = P0;
  float* x1   = P0 + NB;
  float* aggr = P0 + 2 * NB;
  float* lin  = P0 + 3 * NB;
  float* hn   = P0 + 4 * NB;
  float* pix  = P0 + 5 * NB;
  gather_pix<<<cdiv(NB, 256), 256, 0, stream>>>(P1, I(1), pix);

  // ---------------- GNN level 1 ----------------
  gemm_f16w<128, 0, DD><<<dim3(N1C / 16, 2), 32, 0, stream>>>(
      pix, pix, pw_emb1, nullptr, x0, N1C);

  float* xc = x0; float* xn = x1;
  const _Float16* lw1[2] = { pw_l1a, pw_l1b };
  for (int bi = 0; bi < 2; ++bi) {
    int pb = 25 + bi * 6;   // pos_w1, pos_w2, lin_w, lin_b, bn_g, bn_b
    hipMemsetAsync(aggr, 0, NB * sizeof(float), stream);
    hipMemsetAsync(cnt1, 0, N1C * sizeof(float), stream);
    edge_msg_aggr<<<E1C, 128, 0, stream>>>(xc, ei1, ew1, F(pb), F(pb + 1),
                                           aggr, cnt1, E1C);
    seg_div<<<cdiv(NB, 256), 256, 0, stream>>>(aggr, cnt1, N1C);
    gemm_f16w<128, 128, DD><<<dim3(N1C / 16, 2), 32, 0, stream>>>(
        xc, aggr, lw1[bi], F(pb + 3), lin, N1C);
    rownorm<<<N1C, 128, 0, stream>>>(lin, hn);
    colstats<<<128, 256, 0, stream>>>(hn, mean, istd, N1C);
    bn1d_apply_relu_res<<<cdiv(NB, 256), 256, 0, stream>>>(
        hn, mean, istd, F(pb + 4), F(pb + 5), xc, xn, NB);
    float* t = xc; xc = xn; xn = t;
  }

  // ---------------- pool nodes -> superpixels ----------------
  float* sp   = P1;
  float* y0   = P1 + NB2;
  float* y1   = P1 + 2 * NB2;
  float* ag2  = P1 + 3 * NB2;
  float* lin2 = P1 + 4 * NB2;
  float* hn2  = P1 + 5 * NB2;
  float* cnt2 = P1 + 6 * NB2;
  float* hg   = P1 + 6 * NB2 + 4096;

  hipMemsetAsync(sp, 0, NB2 * sizeof(float), stream);
  hipMemsetAsync(cnt2, 0, N2C * sizeof(float), stream);
  seg_accum<<<N1C, 128, 0, stream>>>(xc, batch1, sp, cnt2);
  seg_div<<<cdiv(NB2, 256), 256, 0, stream>>>(sp, cnt2, N2C);

  // ---------------- GNN level 2 ----------------
  gemm_f16w<128, 0, DD><<<dim3(N2C / 16, 2), 32, 0, stream>>>(
      sp, sp, pw_emb2, nullptr, y0, N2C);

  float* yc = y0; float* yn = y1;
  for (int bi = 0; bi < 4; ++bi) {
    int pb = 38 + bi * 6;
    hipMemsetAsync(ag2, 0, NB2 * sizeof(float), stream);
    hipMemsetAsync(cnt2, 0, N2C * sizeof(float), stream);
    edge_msg_aggr<<<E2C, 128, 0, stream>>>(yc, ei2, ew2, F(pb), F(pb + 1),
                                           ag2, cnt2, E2C);
    seg_div<<<cdiv(NB2, 256), 256, 0, stream>>>(ag2, cnt2, N2C);
    gemm_f16w<128, 128, DD><<<dim3(N2C / 16, 2), 32, 0, stream>>>(
        yc, ag2, pw_l2[bi], F(pb + 3), lin2, N2C);
    rownorm<<<N2C, 128, 0, stream>>>(lin2, hn2);
    colstats<<<128, 256, 0, stream>>>(hn2, mean, istd, N2C);
    bn1d_apply_relu_res<<<cdiv(NB2, 256), 256, 0, stream>>>(
        hn2, mean, istd, F(pb + 4), F(pb + 5), yc, yn, NB2);
    float* t = yc; yc = yn; yn = t;
  }

  // ---------------- attention + segment max + readout ----------------
  attn_scale<<<N2C, 128, 0, stream>>>(yc, F(62));
  init_val<<<cdiv(BB * DD, 256), 256, 0, stream>>>(hg, BB * DD, -INFINITY);
  seg_max<<<N2C, 128, 0, stream>>>(yc, batch2, hg);
  gemm_f16w<128, 0, 1000><<<dim3(1, 16), 32, 0, stream>>>(
      hg, hg, pw_ro, nullptr, (float*)d_out, BB);
}
